// Retention_78554951844356
// MI455X (gfx1250) — compile-verified, matrix operand order checked by reading, not run on hardware
//
#include <hip/hip_runtime.h>

// Problem constants (match the reference)
#define Bdim 4
#define Sdim 4096
#define Hdim 1024
// log2(0.96875)
#define L2GAMMA (-0.045804051f)

typedef __attribute__((ext_vector_type(16))) __bf16 v16bf;
typedef __attribute__((ext_vector_type(8)))  __bf16 v8bf;
typedef __attribute__((ext_vector_type(8)))  float  v8f;

// ---------------------------------------------------------------------------
// WMMA helper: D = A(16x32 bf16) * B(32x16 bf16) + C(16x16 f32)
// ---------------------------------------------------------------------------
__device__ static inline v8f wmma_bf16(v16bf a, v16bf b, v8f c) {
  return __builtin_amdgcn_wmma_f32_16x16x32_bf16(
      /*neg_a=*/false, a, /*neg_b=*/false, b,
      /*c_mod=*/(short)0, c, /*reuse_a=*/false, /*reuse_b=*/false);
}

// ---------------------------------------------------------------------------
// CDNA5 async DMA: copy 16B global -> LDS per lane, tracked by ASYNCcnt.
// GV mode: 64-bit vaddr pair, no saddr.  LDS destination address is the low
// 32 bits of the generic (flat) shared pointer (flat-LDS aperture mapping).
// ---------------------------------------------------------------------------
__device__ static inline void async_copy16(void* lds_dst, const void* gsrc) {
  asm volatile("global_load_async_to_lds_b128 %0, %1, off"
               :
               : "v"((unsigned int)(size_t)lds_dst), "v"(gsrc)
               : "memory");
}
__device__ static inline void wait_async0() {
  asm volatile("s_wait_asynccnt 0x0" ::: "memory");
}

// ---------------------------------------------------------------------------
// Fragment loader for a 16x32 bf16 tile stored row-major with `stride` elems.
// Matches CDNA5 ISA layout (CDNA5 ISA §7.12.2):
//   lanes 0-15 : row = lane,      regs 0-3 = K0..7,  regs 4-7 = K16..23
//   lanes 16-31: row = lane-16,   regs 0-3 = K8..15, regs 4-7 = K24..31
// Works for A operands (row = M) and B operands loaded from a [N][K] tile
// (row = N).  Two aligned 16-byte loads per lane.
// ---------------------------------------------------------------------------
__device__ static inline v16bf load_frag(const __bf16* __restrict__ base,
                                         int stride) {
  const int lane = threadIdx.x & 31;
  const int row  = lane & 15;
  const int hi   = lane >> 4;          // 0 or 1
  const __bf16* p = base + row * stride + hi * 8;
  union { v16bf v; v8bf h[2]; } u;
  u.h[0] = *(const v8bf*)(p);          // K 0..7   (or 8..15)
  u.h[1] = *(const v8bf*)(p + 16);     // K 16..23 (or 24..31)
  return u.v;
}

__device__ static inline v8bf cvt8(float4 a, float4 b) {
  v8bf r;
  r[0] = (__bf16)a.x; r[1] = (__bf16)a.y; r[2] = (__bf16)a.z; r[3] = (__bf16)a.w;
  r[4] = (__bf16)b.x; r[5] = (__bf16)b.y; r[6] = (__bf16)b.z; r[7] = (__bf16)b.w;
  return r;
}

// ---------------------------------------------------------------------------
// Projection kernel:  q = X*Wq^T + bq ; k = X*Wk^T + bk ; v = X*Wv^T
//   X: [B*S, H] f32      W*: [H(out), H(in)] f32
//   q,k stored row-major bf16 [B*S, H]; v stored TRANSPOSED bf16 [B, H, S]
// blockIdx.z selects which projection.  Workgroup tile: 64 rows x 128 cols,
// 8 waves as 4(m) x 2(n); each wave owns 16x64 (4 wmma accumulators).
// (f32->bf16 conversion happens while staging tiles, so the DMA path is not
//  usable here; plain loads + v_cvt_pk_bf16_f32 + ds stores.)
// ---------------------------------------------------------------------------
__global__ __launch_bounds__(256) void proj_kernel(
    const float* __restrict__ x,
    const float* __restrict__ Wq, const float* __restrict__ bq,
    const float* __restrict__ Wk, const float* __restrict__ bk,
    const float* __restrict__ Wv,
    __bf16* __restrict__ qb, __bf16* __restrict__ kb,
    __bf16* __restrict__ vt) {
  __shared__ __align__(16) __bf16 Xs[64 * 32];
  __shared__ __align__(16) __bf16 Ws[128 * 32];

  const int mode  = blockIdx.z;                 // 0=q 1=k 2=v
  const float* W    = (mode == 0) ? Wq : (mode == 1) ? Wk : Wv;
  const float* bias = (mode == 0) ? bq : (mode == 1) ? bk : nullptr;

  const int mBase = blockIdx.y * 64;            // over B*S = 16384
  const int nBase = blockIdx.x * 128;           // over H
  const int tid   = threadIdx.x;
  const int wave  = tid >> 5;
  const int lane  = tid & 31;
  const int wm    = wave >> 1;                  // 0..3
  const int wn    = wave & 1;                   // 0..1

  v8f acc[4] = {};

  for (int k0 = 0; k0 < Hdim; k0 += 32) {
    // Stage X tile (64x32) with f32->bf16 conversion
    {
      const int row = tid >> 2;
      const int c8  = (tid & 3) * 8;
      const float4* src =
          (const float4*)(x + (size_t)(mBase + row) * Hdim + k0 + c8);
      *(v8bf*)(&Xs[row * 32 + c8]) = cvt8(src[0], src[1]);
    }
    // Stage W tile (128x32) with conversion
    {
      const int row = tid >> 1;
      const int c16 = (tid & 1) * 16;
      const float4* src =
          (const float4*)(W + (size_t)(nBase + row) * Hdim + k0 + c16);
      *(v8bf*)(&Ws[row * 32 + c16])     = cvt8(src[0], src[1]);
      *(v8bf*)(&Ws[row * 32 + c16 + 8]) = cvt8(src[2], src[3]);
    }
    __syncthreads();

    const v16bf a = load_frag(&Xs[(wm * 16) * 32], 32);
#pragma unroll
    for (int nt = 0; nt < 4; ++nt) {
      const v16bf bfr = load_frag(&Ws[(wn * 64 + nt * 16) * 32], 32);
      acc[nt] = wmma_bf16(a, bfr, acc[nt]);
    }
    __syncthreads();
  }

  // Epilogue: bias + convert + store (v transposed)
  const int nlo = lane & 15;
  const int mhi = (lane >> 4) * 8;
#pragma unroll
  for (int nt = 0; nt < 4; ++nt) {
    const int n = nBase + wn * 64 + nt * 16 + nlo;
    const float badd = (mode < 2) ? bias[n] : 0.0f;
#pragma unroll
    for (int r = 0; r < 8; ++r) {
      const int m = mBase + wm * 16 + mhi + r;
      const float val = acc[nt][r] + badd;
      if (mode == 0) {
        qb[(size_t)m * Hdim + n] = (__bf16)val;
      } else if (mode == 1) {
        kb[(size_t)m * Hdim + n] = (__bf16)val;
      } else {
        const int bidx = m >> 12;          // m / S
        const int s    = m & (Sdim - 1);   // m % S
        vt[((size_t)bidx * Hdim + n) * Sdim + s] = (__bf16)val;
      }
    }
  }
}

// ---------------------------------------------------------------------------
// Retention attention kernel (causal gamma-decay):
//   scores = (Q K^T) * gamma^(i-j)  (i>=j),   O = scores V
// One workgroup per (batch, 32-query tile); 8 waves, each owning a 128-wide
// h-chunk (for the Q.K^T contraction) and 128 output columns (for S.V).
// K tile (32x1024) and V^T tile (1024x32) are DMA'd into LDS with CDNA5
// GLOBAL_LOAD_ASYNC_TO_LDS_B128 (ASYNCcnt / s_wait_asynccnt).  Two 134KB
// workgroups co-reside per 320KB WGP so staging overlaps the other
// workgroup's WMMA phase.  Cross-wave score reduction uses ds_add_f32.
// ---------------------------------------------------------------------------
__global__ __launch_bounds__(256) void retn_kernel(
    const __bf16* __restrict__ qb, const __bf16* __restrict__ kb,
    const __bf16* __restrict__ vt, float* __restrict__ out) {
  __shared__ __align__(16) __bf16 Ks[32 * 1024];    // 64 KB  [key][h]
  __shared__ __align__(16) __bf16 Vts[1024 * 32];   // 64 KB  [col][key]
  __shared__ float  Sred[32 * 32];                  // 4 KB   score reduction
  __shared__ __align__(16) __bf16 Sbf[32 * 32];     // 2 KB   decayed scores

  const int b     = blockIdx.y;
  const int qTile = blockIdx.x;
  const int qBase = qTile * 32;
  const int tid   = threadIdx.x;
  const int wave  = tid >> 5;
  const int lane  = tid & 31;
  const int hChunk = wave * 128;                    // this wave's h / col range

  // Q fragments held in registers: 2 m-tiles x 4 k-steps over this h-chunk
  v16bf qf[2][4];
#pragma unroll
  for (int mt = 0; mt < 2; ++mt)
#pragma unroll
    for (int ks = 0; ks < 4; ++ks)
      qf[mt][ks] = load_frag(
          qb + ((size_t)(b * Sdim + qBase + mt * 16)) * Hdim + hChunk + ks * 32,
          Hdim);

  v8f o[2][8] = {};   // 32 rows x 128 cols of output accumulator per wave

  for (int kblk = 0; kblk <= qTile; ++kblk) {
    const int kBase = kblk * 32;

    // --- async-DMA K tile: 32 rows x 1024 bf16 (16 x b128 per thread) ---
    {
      const int row = tid >> 3;
      const int seg = (tid & 7) * 128;   // 128 bf16 = 256B per thread segment
      const __bf16* src =
          kb + ((size_t)(b * Sdim + kBase + row)) * Hdim + seg;
      __bf16* dst = &Ks[row * 1024 + seg];
#pragma unroll
      for (int i = 0; i < 16; ++i)
        async_copy16(dst + i * 8, src + i * 8);
    }
    // --- async-DMA V^T tile: 1024 rows(cols) x 32 keys ---
    {
      const int col0 = tid * 4;
#pragma unroll
      for (int c = 0; c < 4; ++c) {
        const __bf16* src =
            vt + ((size_t)(b * Hdim + col0 + c)) * Sdim + kBase;
        __bf16* dst = &Vts[(col0 + c) * 32];
        async_copy16(dst,      src);
        async_copy16(dst + 8,  src + 8);
        async_copy16(dst + 16, src + 16);
        async_copy16(dst + 24, src + 24);
      }
    }
    // --- prefetch next key block toward L2/WGP$ while this one is consumed ---
    if (kblk < qTile) {
      __builtin_prefetch(
          kb + ((size_t)(b * Sdim + kBase + 32)) * Hdim + (size_t)tid * 128,
          0, 1);
      __builtin_prefetch(
          vt + ((size_t)b * Hdim + tid * 4) * Sdim + kBase + 32, 0, 1);
    }
    // --- zero the score-reduction buffer ---
#pragma unroll
    for (int i = 0; i < 4; ++i) Sred[tid * 4 + i] = 0.0f;

    wait_async0();       // this wave's DMA into LDS complete
    __syncthreads();     // all waves' tiles visible

    // --- partial scores over this wave's h-chunk: S[32x32] ---
    v8f sacc[2][2] = {};
#pragma unroll
    for (int ks = 0; ks < 4; ++ks) {
      const v16bf kf0 = load_frag(&Ks[hChunk + ks * 32], 1024);             // keys 0..15
      const v16bf kf1 = load_frag(&Ks[16 * 1024 + hChunk + ks * 32], 1024); // keys 16..31
#pragma unroll
      for (int mt = 0; mt < 2; ++mt) {
        sacc[mt][0] = wmma_bf16(qf[mt][ks], kf0, sacc[mt][0]);
        sacc[mt][1] = wmma_bf16(qf[mt][ks], kf1, sacc[mt][1]);
      }
    }
    // cross-wave reduction via LDS float atomics
    {
      const int nlo = lane & 15;
      const int mhi = (lane >> 4) * 8;
#pragma unroll
      for (int mt = 0; mt < 2; ++mt)
#pragma unroll
        for (int nt = 0; nt < 2; ++nt)
#pragma unroll
          for (int r = 0; r < 8; ++r)
            atomicAdd(&Sred[(mt * 16 + mhi + r) * 32 + nt * 16 + nlo],
                      sacc[mt][nt][r]);
    }
    __syncthreads();

    // --- apply causal gamma decay, convert to bf16 ---
#pragma unroll
    for (int i = 0; i < 4; ++i) {
      const int idx = tid * 4 + i;
      const int m = idx >> 5;
      const int n = idx & 31;
      const int e = (qBase + m) - (kBase + n);
      const float w = (e >= 0) ? exp2f((float)e * L2GAMMA) : 0.0f;
      Sbf[idx] = (__bf16)(Sred[idx] * w);
    }
    __syncthreads();

    // --- O += S * V over this wave's 128 output columns ---
    {
      const v16bf af0 = load_frag(&Sbf[0], 32);
      const v16bf af1 = load_frag(&Sbf[16 * 32], 32);
#pragma unroll
      for (int nt = 0; nt < 8; ++nt) {
        const v16bf vf = load_frag(&Vts[(hChunk + nt * 16) * 32], 32);
        o[0][nt] = wmma_bf16(af0, vf, o[0][nt]);
        o[1][nt] = wmma_bf16(af1, vf, o[1][nt]);
      }
    }
    __syncthreads();   // before overwriting Ks/Vts/Sred next iteration
  }

  // --- epilogue: store O (f32) ---
  const int nlo = lane & 15;
  const int mhi = (lane >> 4) * 8;
#pragma unroll
  for (int mt = 0; mt < 2; ++mt)
#pragma unroll
    for (int nt = 0; nt < 8; ++nt)
#pragma unroll
      for (int r = 0; r < 8; ++r) {
        const int m = qBase + mt * 16 + mhi + r;
        const int n = hChunk + nt * 16 + nlo;
        out[((size_t)(b * Sdim + m)) * Hdim + n] = o[mt][nt][r];
      }
}

// ---------------------------------------------------------------------------
extern "C" void kernel_launch(void* const* d_in, const int* in_sizes, int n_in,
                              void* d_out, int out_size, void* d_ws,
                              size_t ws_size, hipStream_t stream) {
  const float* x  = (const float*)d_in[0];
  const float* Wq = (const float*)d_in[1];
  const float* bq = (const float*)d_in[2];
  const float* Wk = (const float*)d_in[3];
  const float* bk = (const float*)d_in[4];
  const float* Wv = (const float*)d_in[5];
  float* out = (float*)d_out;

  const size_t BSH = (size_t)Bdim * Sdim * Hdim;   // 16.7M elems
  __bf16* qb = (__bf16*)d_ws;                      // 32 MB
  __bf16* kb = qb + BSH;                           // 32 MB
  __bf16* vt = kb + BSH;                           // 32 MB (V transposed [B,H,S])

  // Projections (q,k row-major; v transposed) — all via bf16 WMMA
  dim3 pgrid(Hdim / 128, (Bdim * Sdim) / 64, 3);   // (8, 256, 3)
  proj_kernel<<<pgrid, 256, 0, stream>>>(x, Wq, bq, Wk, bk, Wv, qb, kb, vt);

  // Causal decayed attention
  dim3 agrid(Sdim / 32, Bdim);                     // (128, 4)
  retn_kernel<<<agrid, 256, 0, stream>>>(qb, kb, vt, out);
}
